// CS2GQAAttention_62199716380870
// MI455X (gfx1250) — matrix operand order repaired and checked by reading, not compiled, not run on hardware
//
#include <hip/hip_runtime.h>
#include <hip/hip_bf16.h>

// ---------------------------------------------------------------------------
// CS2 GQA attention with 2D RoPE for gfx1250 (MI455X), bf16 WMMA pipeline.
//   B=4 L=896 D=2048 HQ=32 HKV=8 HD=64 TPF=7
// Pipeline:
//   cvt  : f32 -> bf16 for x, wq, wk, wv, wo
//   gemm : Q = rope(x @ wq^T), K = rope(x @ wk^T), V = x @ wv^T   (bf16 out)
//          -> LDS-staged, double-buffered, GLOBAL_LOAD_ASYNC_TO_LDS_B128
//   attn : flash attention per (b, hq, 16-query tile), GQA g=4    (bf16 out)
//   gemm : out = attn_out @ wo^T                                  (f32 -> d_out)
// All matmuls use v_wmma_f32_16x16x32_bf16.
// ---------------------------------------------------------------------------

typedef __attribute__((ext_vector_type(16))) __bf16 v16bf;
typedef __attribute__((ext_vector_type(8)))  __bf16 v8bf;
typedef __attribute__((ext_vector_type(8)))  float  v8f;
typedef int v4i_vs __attribute__((vector_size(4 * sizeof(int))));  // builtin pointee

#define B_    4
#define L_    896
#define D_    2048
#define HQ_   32
#define HKV_  8
#define HD_   64
#define DKV_  512
#define TPF_  7
#define MROWS (B_ * L_)          // 3584

#define WMMA_BF16(a, b, c) \
  __builtin_amdgcn_wmma_f32_16x16x32_bf16(false, (a), false, (b), (short)0, (c), false, false)

// ---- async-to-LDS path (CDNA5), guarded so either toolchain compiles -------
#if defined(__has_builtin)
#if __has_builtin(__builtin_amdgcn_global_load_async_to_lds_b128)
#define HAS_ASYNC_LDS 1
#endif
#endif
#ifndef HAS_ASYNC_LDS
#define HAS_ASYNC_LDS 0
#endif

#define AS1 __attribute__((address_space(1)))
#define AS3 __attribute__((address_space(3)))

static __device__ __forceinline__ void async_cp16(const __bf16* g, __bf16* l) {
#if HAS_ASYNC_LDS
  __builtin_amdgcn_global_load_async_to_lds_b128(
      (AS1 v4i_vs*)g, (AS3 v4i_vs*)l, 0, 0);
#else
  *(v8bf*)l = *(const v8bf*)g;
#endif
}

#if HAS_ASYNC_LDS
#if __has_builtin(__builtin_amdgcn_s_wait_asynccnt)
#define WAIT_ASYNC(n) __builtin_amdgcn_s_wait_asynccnt(n)
#else
#define WAIT_ASYNC(n) asm volatile("s_wait_asynccnt " #n ::: "memory")
#endif
#else
#define WAIT_ASYNC(n) ((void)0)
#endif

static __device__ __forceinline__ int lane_id() { return (int)(threadIdx.x & 31); }

// A fragment: 16x32 bf16, row-major source with leading dim `ld` (elements).
// Contiguous-k order used here (not the ISA table order): lane = 16h + m; lane holds K in [16h, 16h+16),
// i.e. one contiguous 32B run -> two b128 loads (global or ds).
static __device__ __forceinline__ v16bf load_a_frag(const __bf16* src, int ld) {
  int lane = lane_id();
  int m = lane & 15, h = lane >> 4;
  const __bf16* p = src + (size_t)m * ld + h * 16;
  v8bf lo = *(const v8bf*)p;          // K = 16h .. 16h+7
  v8bf hi = *(const v8bf*)(p + 8);    // K = 16h+8 .. 16h+15
  return __builtin_shufflevector(lo, hi, 0, 1, 2, 3, 4, 5, 6, 7,
                                          8, 9, 10, 11, 12, 13, 14, 15);
}

// B fragment: logical 32x16 (K x N), sourced from an N x K row-major matrix
// (element(k,n) = src[n*ld + k]). Lane (n, h) holds K = 16h..16h+15: one 32B run.
static __device__ __forceinline__ v16bf load_b_frag_nk(const __bf16* src, int ld) {
  int lane = lane_id();
  int n = lane & 15, h = lane >> 4;
  return *(const v16bf*)(src + (size_t)n * ld + h * 16);
}

// B fragment from a K x N row-major source (element(k,n) = src[k*ld + n]).
static __device__ __forceinline__ v16bf load_b_frag_kn(const __bf16* src, int ld) {
  int lane = lane_id();
  int n = lane & 15, h = lane >> 4;
  v16bf b;
#pragma unroll
  for (int j = 0; j < 8; ++j) {
    int k = h * 16 + 2 * j;
    b[2 * j]     = src[(size_t)k * ld + n];
    b[2 * j + 1] = src[(size_t)(k + 1) * ld + n];
  }
  return b;
}

// ---------------------------------------------------------------------------
__global__ void cvt_f32_bf16(const float* __restrict__ src,
                             __bf16* __restrict__ dst, int n) {
  int i = blockIdx.x * blockDim.x + threadIdx.x;
  if (i < n) dst[i] = (__bf16)src[i];
}

// ---------------------------------------------------------------------------
// C[M x N] = A[M x K] (bf16) @ W[N x K]^T (bf16), f32 accumulate.
// Block: 256 threads = 8 waves as 4(m) x 2(n); block tile 64 x 128,
// wave tile 16 x 64. K staged through LDS in 64-wide slabs, double buffered,
// copies issued with GLOBAL_LOAD_ASYNC_TO_LDS_B128 when the toolchain has it.
// Optional fused 2D-RoPE epilogue (pair partner lives in lane^1 -> shfl_xor).
#define KTILE  64
#define LDS_LD 80   // 64 + 16 pad: 160B row stride (32B-aligned frag reads)

__global__ __launch_bounds__(256) void gemm_bf16_wmma(
    const __bf16* __restrict__ A, const __bf16* __restrict__ W,
    __bf16* __restrict__ Cb, float* __restrict__ Cf,
    int M, int N, int K, int applyRope) {
  __shared__ __align__(16) __bf16 ldsA[2][64 * LDS_LD];    // 20 KB
  __shared__ __align__(16) __bf16 ldsB[2][128 * LDS_LD];   // 40 KB

  int tid  = threadIdx.x;
  int wave = tid >> 5;
  int wm = wave & 3, wn = wave >> 2;
  int mblk = blockIdx.y * 64;
  int nblk = blockIdx.x * 128;
  int m0 = mblk + wm * 16;
  int n0 = nblk + wn * 64;

  const __bf16* gA = A + (size_t)mblk * K;   // 64 x K slab
  const __bf16* gB = W + (size_t)nblk * K;   // 128 x K slab

  // Stage copy: A 64x64 el (512 x 16B segs, 2/thread), B 128x64 (1024 segs, 4/thread)
  auto copy_stage = [&](int buf, int k0) {
#pragma unroll
    for (int t = 0; t < 2; ++t) {
      int idx = tid + t * 256;
      int row = idx >> 3, seg = idx & 7;
      async_cp16(gA + (size_t)row * K + k0 + seg * 8,
                 &ldsA[buf][row * LDS_LD + seg * 8]);
    }
#pragma unroll
    for (int t = 0; t < 4; ++t) {
      int idx = tid + t * 256;
      int row = idx >> 3, seg = idx & 7;
      async_cp16(gB + (size_t)row * K + k0 + seg * 8,
                 &ldsB[buf][row * LDS_LD + seg * 8]);
    }
  };

  v8f acc[4] = {};
  const int NT = K / KTILE;   // 32
  copy_stage(0, 0);
  for (int kt = 0; kt < NT; ++kt) {
    int cur = kt & 1;
    if (kt + 1 < NT) {
      copy_stage(cur ^ 1, (kt + 1) * KTILE);
      WAIT_ASYNC(6);   // 6 newest (next stage) in flight; current stage drained
    } else {
      WAIT_ASYNC(0);
    }
    __syncthreads();
    const __bf16* lA = &ldsA[cur][(size_t)(wm * 16) * LDS_LD];
#pragma unroll
    for (int ks = 0; ks < 2; ++ks) {
      v16bf a = load_a_frag(lA + ks * 32, LDS_LD);
#pragma unroll
      for (int t = 0; t < 4; ++t) {
        v16bf b = load_b_frag_nk(
            &ldsB[cur][(size_t)(wn * 64 + 16 * t) * LDS_LD] + ks * 32, LDS_LD);
        acc[t] = WMMA_BF16(a, b, acc[t]);
      }
    }
    __syncthreads();
  }

  int lane = lane_id();
  int nn = lane & 15, h = lane >> 4;
#pragma unroll
  for (int t = 0; t < 4; ++t) {
    int n = n0 + t * 16 + nn;
#pragma unroll
    for (int r = 0; r < 8; ++r) {
      int m = m0 + h * 8 + r;
      float x = acc[t][r];
      float partner = __shfl_xor(x, 1, 32);   // rotary pair partner (n ^ 1)
      if (applyRope) {
        int d   = n & (HD_ - 1);              // dim within head
        int l   = m % L_;                     // token index
        int pos = (d < 32) ? (l / TPF_) : (l % TPF_);
        int p   = (d & 31) >> 1;              // frequency index 0..15
        // inv_freq = 10000^(-p/16) = exp(-p * ln(10000)/16)
        float ang = (float)pos * __expf(-0.5756462732485115f * (float)p);
        float s, c;
        __sincosf(ang, &s, &c);
        x = (d & 1) ? (partner * s + x * c) : (x * c - partner * s);
      }
      if (Cf) Cf[(size_t)m * N + n] = x;
      else    Cb[(size_t)m * N + n] = (__bf16)x;
    }
  }
}

// ---------------------------------------------------------------------------
// Flash attention: one wave per (batch, q-head, 16-query tile).
// S = Q K^T via 4 WMMAs / 32-key tile; P re-swizzled C->A layout through LDS;
// O += P V via 4 WMMAs. Online softmax with shfl_xor row reductions.
__global__ __launch_bounds__(128) void attn_fwd_wmma(
    const __bf16* __restrict__ Q, const __bf16* __restrict__ Kt,
    const __bf16* __restrict__ Vt, __bf16* __restrict__ O) {
  __shared__ __align__(16) __bf16 p_lds[4][16 * 32];

  int wave = threadIdx.x >> 5;
  int lane = lane_id();
  int nn = lane & 15, h = lane >> 4;
  int qt  = blockIdx.x * 4 + wave;  // 0..55
  int hq  = blockIdx.y;             // 0..31
  int b   = blockIdx.z;             // 0..3
  int hkv = hq >> 2;                // GQA group of 4
  int m0  = qt * 16;

  const __bf16* qptr = Q + (size_t)(b * L_ + m0) * D_ + hq * HD_;
  v16bf qa0 = load_a_frag(qptr, D_);        // d = 0..31
  v16bf qa1 = load_a_frag(qptr + 32, D_);   // d = 32..63

  v8f o0 = {}, o1 = {}, o2 = {}, o3 = {};
  float mi[8], li[8];
#pragma unroll
  for (int r = 0; r < 8; ++r) { mi[r] = -1e30f; li[r] = 0.0f; }

  __bf16* pl = p_lds[wave];
  const float scale = 0.125f;  // 1/sqrt(64)
  int kend = m0 + 16;

  for (int key0 = 0; key0 < kend; key0 += 32) {
    // --- S = Q K^T : K rows are tokens (N), features are K-dim -> "nk" frags
    const __bf16* kptr = Kt + (size_t)(b * L_ + key0) * DKV_ + hkv * HD_;
    v16bf kb00 = load_b_frag_nk(kptr, DKV_);
    v16bf kb01 = load_b_frag_nk(kptr + 32, DKV_);
    v16bf kb10 = load_b_frag_nk(kptr + (size_t)16 * DKV_, DKV_);
    v16bf kb11 = load_b_frag_nk(kptr + (size_t)16 * DKV_ + 32, DKV_);
    v8f s0 = {}, s1 = {};
    s0 = WMMA_BF16(qa0, kb00, s0);
    s0 = WMMA_BF16(qa1, kb01, s0);
    s1 = WMMA_BF16(qa0, kb10, s1);
    s1 = WMMA_BF16(qa1, kb11, s1);

    // --- causal mask + scale, per-row tile max
    float mloc[8];
#pragma unroll
    for (int r = 0; r < 8; ++r) {
      int qrow = m0 + h * 8 + r;
      float a0 = (key0 + nn      <= qrow) ? s0[r] * scale : -1e30f;
      float a1 = (key0 + 16 + nn <= qrow) ? s1[r] * scale : -1e30f;
      s0[r] = a0; s1[r] = a1;
      mloc[r] = fmaxf(a0, a1);
    }
#pragma unroll
    for (int off = 1; off < 16; off <<= 1) {
#pragma unroll
      for (int r = 0; r < 8; ++r)
        mloc[r] = fmaxf(mloc[r], __shfl_xor(mloc[r], off, 32));
    }

    // --- online softmax update; stage P into LDS in plain row-major
#pragma unroll
    for (int r = 0; r < 8; ++r) {
      float nm    = fmaxf(mi[r], mloc[r]);
      float alpha = __expf(mi[r] - nm);
      mi[r] = nm;
      float p0 = __expf(s0[r] - nm);
      float p1 = __expf(s1[r] - nm);
      float rs = p0 + p1;
#pragma unroll
      for (int off = 1; off < 16; off <<= 1) rs += __shfl_xor(rs, off, 32);
      li[r] = li[r] * alpha + rs;
      pl[(h * 8 + r) * 32 + nn]      = (__bf16)p0;
      pl[(h * 8 + r) * 32 + 16 + nn] = (__bf16)p1;
      o0[r] *= alpha; o1[r] *= alpha; o2[r] *= alpha; o3[r] *= alpha;
    }
    // Cross-lane P exchange through LDS within one wave: DS ops are in-order,
    // but fence the compiler and drain DScnt before re-reading.
    asm volatile("s_wait_dscnt 0" ::: "memory");
    v16bf pa = load_a_frag(pl, 32);  // P as 16x32 A fragment

    // --- O += P V : V is (token=K-dim) x (feature=N) -> "kn" frags
    const __bf16* vptr = Vt + (size_t)(b * L_ + key0) * DKV_ + hkv * HD_;
    o0 = WMMA_BF16(pa, load_b_frag_kn(vptr + 0,  DKV_), o0);
    o1 = WMMA_BF16(pa, load_b_frag_kn(vptr + 16, DKV_), o1);
    o2 = WMMA_BF16(pa, load_b_frag_kn(vptr + 32, DKV_), o2);
    o3 = WMMA_BF16(pa, load_b_frag_kn(vptr + 48, DKV_), o3);
  }

  // --- finalize: O /= l, write bf16 into (B*L, HQ*HD) layout
  __bf16* optr = O + (size_t)(b * L_ + m0) * D_ + hq * HD_;
#pragma unroll
  for (int r = 0; r < 8; ++r) {
    float inv = 1.0f / li[r];
    size_t row = (size_t)(h * 8 + r) * D_;
    optr[row + nn]      = (__bf16)(o0[r] * inv);
    optr[row + 16 + nn] = (__bf16)(o1[r] * inv);
    optr[row + 32 + nn] = (__bf16)(o2[r] * inv);
    optr[row + 48 + nn] = (__bf16)(o3[r] * inv);
  }
}

// ---------------------------------------------------------------------------
extern "C" void kernel_launch(void* const* d_in, const int* in_sizes, int n_in,
                              void* d_out, int out_size, void* d_ws, size_t ws_size,
                              hipStream_t stream) {
  (void)in_sizes; (void)n_in; (void)out_size; (void)ws_size;
  const float* x  = (const float*)d_in[0];
  const float* wq = (const float*)d_in[1];
  const float* wk = (const float*)d_in[2];
  const float* wv = (const float*)d_in[3];
  const float* wo = (const float*)d_in[4];
  // d_in[5]/d_in[6] (temporal/structural pos) are l/7 and l%7 — computed inline.

  char* ws = (char*)d_ws;
  size_t off = 0;
  auto take = [&](size_t bytes) {
    void* p = ws + off;
    off += (bytes + 255) & ~(size_t)255;
    return p;
  };
  __bf16* xb  = (__bf16*)take((size_t)MROWS * D_ * 2);
  __bf16* wqb = (__bf16*)take((size_t)D_ * D_ * 2);
  __bf16* wkb = (__bf16*)take((size_t)DKV_ * D_ * 2);
  __bf16* wvb = (__bf16*)take((size_t)DKV_ * D_ * 2);
  __bf16* wob = (__bf16*)take((size_t)D_ * D_ * 2);
  __bf16* Qb  = (__bf16*)take((size_t)MROWS * D_ * 2);
  __bf16* Kb  = (__bf16*)take((size_t)MROWS * DKV_ * 2);
  __bf16* Vb  = (__bf16*)take((size_t)MROWS * DKV_ * 2);
  __bf16* Ab  = (__bf16*)take((size_t)MROWS * D_ * 2);

  // 1) f32 -> bf16 conversions
  auto cvt = [&](const float* s, __bf16* d, int n) {
    cvt_f32_bf16<<<(n + 255) / 256, 256, 0, stream>>>(s, d, n);
  };
  cvt(x,  xb,  MROWS * D_);
  cvt(wq, wqb, D_ * D_);
  cvt(wk, wkb, DKV_ * D_);
  cvt(wv, wvb, DKV_ * D_);
  cvt(wo, wob, D_ * D_);

  dim3 blk(256);
  // 2) Q/K/V projections (RoPE fused into Q and K epilogues)
  gemm_bf16_wmma<<<dim3(D_ / 128,   MROWS / 64), blk, 0, stream>>>(
      xb, wqb, Qb, nullptr, MROWS, D_,   D_, 1);
  gemm_bf16_wmma<<<dim3(DKV_ / 128, MROWS / 64), blk, 0, stream>>>(
      xb, wkb, Kb, nullptr, MROWS, DKV_, D_, 1);
  gemm_bf16_wmma<<<dim3(DKV_ / 128, MROWS / 64), blk, 0, stream>>>(
      xb, wvb, Vb, nullptr, MROWS, DKV_, D_, 0);

  // 3) attention: grid (14 q-tile groups of 4 waves, 32 heads, 4 batches)
  attn_fwd_wmma<<<dim3((L_ / 16) / 4, HQ_, B_), dim3(128), 0, stream>>>(
      Qb, Kb, Vb, Ab);

  // 4) output projection -> f32 d_out
  gemm_bf16_wmma<<<dim3(D_ / 128, MROWS / 64), blk, 0, stream>>>(
      Ab, wob, nullptr, (float*)d_out, MROWS, D_, D_, 0);
}
